// CTPNet_62130996903979
// MI455X (gfx1250) — matrix operand order repaired and checked by reading, not compiled, not run on hardware
//
#include <hip/hip_runtime.h>
#include <hip/hip_bf16.h>
#include <stdint.h>

// Problem sizes (fixed by the reference)
constexpr int Bsz  = 8192;
constexpr int Gsz  = 20000;
constexpr int H1sz = 512;
constexpr int H2sz = 256;
constexpr int Psz  = 128;
constexpr int Asz  = 64;

typedef __attribute__((ext_vector_type(16))) __bf16 v16bf;
typedef __attribute__((ext_vector_type(8)))  float  v8f;

// 16 x bf16 fragment = 8 VGPRs; assembled from two 16B chunks.
union FragAB {
    v16bf   v;
    uint32_t u[8];
    uint4    q[2];
};

// f32 pair -> packed bf16 (round-half-up): 2x v_add + 1x v_perm_b32.
__device__ __forceinline__ uint32_t pack2_bf16(float a, float b) {
    uint32_t ua = __float_as_uint(a) + 0x8000u;
    uint32_t ub = __float_as_uint(b) + 0x8000u;
    // result byte0,1 = ua bytes 2,3 (sel 0x02,0x03 from src1); byte2,3 = ub bytes 2,3
    return __builtin_amdgcn_perm(ub, ua, 0x07060302u);
}

__device__ __forceinline__ uint16_t f2bf_bits(float a) {
    return (uint16_t)((__float_as_uint(a) + 0x8000u) >> 16);
}

__device__ __forceinline__ v8f wmma_bf16(const FragAB& a, const FragAB& b, v8f c) {
    return __builtin_amdgcn_wmma_f32_16x16x32_bf16(
        /*neg_a=*/false, a.v, /*neg_b=*/false, b.v,
        /*c_mod=*/(short)0, c, /*reuse_a=*/false, /*reuse_b=*/false);
}

// CDNA5 async global->LDS copy (16B), tracked by ASYNCcnt (ISA §15.18.3 op 98).
__device__ __forceinline__ void async_copy_b128(uint32_t lds_byte_addr, const float* gsrc) {
    asm volatile("global_load_async_to_lds_b128 %0, %1, off"
                 :: "v"(lds_byte_addr), "v"(gsrc)
                 : "memory");
}
__device__ __forceinline__ void wait_asynccnt0() {
    asm volatile("s_wait_asynccnt 0x0" ::: "memory");
}

// ---------------------------------------------------------------- prep kernels

__global__ void k_cvt_bf16(const float* __restrict__ in, uint16_t* __restrict__ out, int n) {
    int i = blockIdx.x * blockDim.x + threadIdx.x;
    int stride = gridDim.x * blockDim.x;
    for (; i < n; i += stride) out[i] = f2bf_bits(in[i]);
}

// W3 [P, H2, A] (f32) -> W3T [P, A, H2] (bf16): K (=H2) becomes contiguous per column.
__global__ void k_w3_transpose(const float* __restrict__ w3, uint16_t* __restrict__ w3t) {
    int i = blockIdx.x * 256 + threadIdx.x;           // over P*A*H2 = 2,097,152
    if (i >= Psz * Asz * H2sz) return;
    int h = i & (H2sz - 1);
    int a = (i >> 8) & (Asz - 1);
    int p = i >> 14;
    w3t[i] = f2bf_bits(w3[((size_t)p * H2sz + h) * Asz + a]);
}

// ---------------------------------------------------------------- GEMM1
// h1 = relu(x @ W1^T + b1), M=8192 N=512 K=20000.
// WG tile: 64(M) x 512(N), 512 threads = 16 waves (4/SIMD for latency hiding);
// wave tile 32x64 (2x4 WMMA tiles, 64 acc VGPRs).
// x (f32): double-buffered LDS staging via GLOBAL_LOAD_ASYNC_TO_LDS_B128 (ASYNCcnt) —
// the DMA runs while the wave does WMMAs; f32->bf16 pack (3 VALU/pair, co-executes
// with XDL WMMA) happens at fragment build. W1 (bf16, L2): immediate-offset b128 loads.
__global__ __launch_bounds__(512) void k_gemm1(
    const float* __restrict__ x, const uint16_t* __restrict__ w1b,
    const float* __restrict__ b1, uint16_t* __restrict__ h1b)
{
    constexpr int KSTEP = 160;               // 20000 = 160 * 125
    constexpr int LDAF  = 164;               // padded LDS row stride in f32 (656B, 16B mult.)
    constexpr int BUFB  = 64 * LDAF * 4;     // 41,984 B per buffer
    __shared__ __align__(16) float sX[2][64 * LDAF];

    const int tid  = threadIdx.x;
    const int lane = tid & 31;
    const int wv   = tid >> 5;               // 0..15
    const int wm   = (wv & 1) * 32;          // wave M offset in tile
    const int wn   = (wv >> 1) * 64;         // wave N offset in tile
    const int half = lane >> 4;              // lane half selects K sub-chunk
    const int l15  = lane & 15;
    const int m0   = blockIdx.x * 64;

    // per-lane B base pointer; fragment addressing uses constant immediates
    const uint16_t* pB0 = w1b + (size_t)(wn + l15) * Gsz + half * 8;

    // staging: 2560 b128 slots per tile / 512 threads = 5 per thread
    uint32_t goff[5];                        // x element offsets
    uint32_t loff[5];                        // LDS byte offsets within one buffer
    #pragma unroll
    for (int i = 0; i < 5; ++i) {
        int idx = tid + i * 512;
        int row = idx / 40;
        int c4  = idx % 40;
        goff[i] = (uint32_t)(m0 + row) * (uint32_t)Gsz + (uint32_t)(c4 * 4);
        loff[i] = (uint32_t)(row * (LDAF * 4) + c4 * 16);
    }
    const uint32_t ldsBase = (uint32_t)(uintptr_t)&sX[0][0];

    const v8f vzero = {0.f,0.f,0.f,0.f,0.f,0.f,0.f,0.f};
    v8f acc[2][4];
    #pragma unroll
    for (int mt = 0; mt < 2; ++mt)
        #pragma unroll
        for (int nt = 0; nt < 4; ++nt) acc[mt][nt] = vzero;

    // prologue: kick off async staging of tile 0 into buffer 0
    #pragma unroll
    for (int i = 0; i < 5; ++i)
        async_copy_b128(ldsBase + loff[i], x + goff[i]);

    int buf = 0;
    for (int k0 = 0; k0 < Gsz; k0 += KSTEP) {
        const bool more = (k0 + KSTEP) < Gsz;
        wait_asynccnt0();                    // my async writes into current buffer done
        __syncthreads();                     // everyone's writes visible; prior reads done

        // issue async staging for NEXT tile into the other buffer (DMA overlaps WMMAs)
        if (more) {
            const uint32_t dstB = ldsBase + (buf ^ 1) * BUFB;
            #pragma unroll
            for (int i = 0; i < 5; ++i)
                async_copy_b128(dstB + loff[i], x + goff[i] + (k0 + KSTEP));
        }

        // compute on current buffer (5 x 8 WMMAs per wave)
        const float* sbuf = &sX[buf][0];
        #pragma unroll
        for (int kk = 0; kk < KSTEP; kk += 32) {
            FragAB afr[2];
            #pragma unroll
            for (int mt = 0; mt < 2; ++mt) {
                const float* pA = sbuf + (wm + mt * 16 + l15) * LDAF + kk + half * 8;
                const float4 c0 = *(const float4*)pA;          // K base+0..3
                const float4 c1 = *(const float4*)(pA + 4);    // K base+4..7
                const float4 c2 = *(const float4*)(pA + 16);   // K base+16..19
                const float4 c3 = *(const float4*)(pA + 20);   // K base+20..23
                afr[mt].u[0] = pack2_bf16(c0.x, c0.y);
                afr[mt].u[1] = pack2_bf16(c0.z, c0.w);
                afr[mt].u[2] = pack2_bf16(c1.x, c1.y);
                afr[mt].u[3] = pack2_bf16(c1.z, c1.w);
                afr[mt].u[4] = pack2_bf16(c2.x, c2.y);
                afr[mt].u[5] = pack2_bf16(c2.z, c2.w);
                afr[mt].u[6] = pack2_bf16(c3.x, c3.y);
                afr[mt].u[7] = pack2_bf16(c3.z, c3.w);
            }
            FragAB bfr[4];
            #pragma unroll
            for (int nt = 0; nt < 4; ++nt) {
                const uint16_t* pB = pB0 + (size_t)(nt * 16) * Gsz + (k0 + kk);
                bfr[nt].q[0] = *(const uint4*)pB;
                bfr[nt].q[1] = *(const uint4*)(pB + 16);
            }
            #pragma unroll
            for (int nt = 0; nt < 4; ++nt) {
                acc[0][nt] = wmma_bf16(afr[0], bfr[nt], acc[0][nt]);
                acc[1][nt] = wmma_bf16(afr[1], bfr[nt], acc[1][nt]);
            }
        }
        buf ^= 1;
    }

    // epilogue: +b1, ReLU, store bf16 (C layout: N=l15, M=i+8*half per VGPR i)
    #pragma unroll
    for (int nt = 0; nt < 4; ++nt) {
        const int n = wn + nt * 16 + l15;
        const float bias = b1[n];
        #pragma unroll
        for (int mt = 0; mt < 2; ++mt)
            #pragma unroll
            for (int i = 0; i < 8; ++i) {
                const int m = m0 + wm + mt * 16 + half * 8 + i;
                h1b[(size_t)m * H1sz + n] = f2bf_bits(fmaxf(acc[mt][nt][i] + bias, 0.f));
            }
    }
}

// ---------------------------------------------------------------- GEMM2
// h2 = relu(h1 @ W2^T + b2), M=8192 N=256 K=512; everything bf16 in L2, no LDS needed.
__global__ __launch_bounds__(256) void k_gemm2(
    const uint16_t* __restrict__ h1b, const uint16_t* __restrict__ w2b,
    const float* __restrict__ b2, uint16_t* __restrict__ h2b)
{
    const int tid  = threadIdx.x;
    const int lane = tid & 31;
    const int wv   = tid >> 5;
    const int wm   = (wv & 1) * 32;
    const int wn   = (wv >> 1) * 64;
    const int half = lane >> 4;
    const int l15  = lane & 15;
    const int m0   = blockIdx.x * 64;

    const uint16_t* pA0 = h1b + (size_t)(m0 + wm + l15) * H1sz + half * 8;
    const uint16_t* pB0 = w2b + (size_t)(wn + l15) * H1sz + half * 8;

    const v8f vzero = {0.f,0.f,0.f,0.f,0.f,0.f,0.f,0.f};
    v8f acc[2][4];
    #pragma unroll
    for (int mt = 0; mt < 2; ++mt)
        #pragma unroll
        for (int nt = 0; nt < 4; ++nt) acc[mt][nt] = vzero;

    #pragma unroll 4
    for (int k0 = 0; k0 < H1sz; k0 += 32) {
        FragAB afr[2], bfr[4];
        #pragma unroll
        for (int mt = 0; mt < 2; ++mt) {
            const uint16_t* pA = pA0 + (size_t)(mt * 16) * H1sz + k0;
            afr[mt].q[0] = *(const uint4*)pA;
            afr[mt].q[1] = *(const uint4*)(pA + 16);
        }
        #pragma unroll
        for (int nt = 0; nt < 4; ++nt) {
            const uint16_t* pB = pB0 + (size_t)(nt * 16) * H1sz + k0;
            bfr[nt].q[0] = *(const uint4*)pB;
            bfr[nt].q[1] = *(const uint4*)(pB + 16);
        }
        #pragma unroll
        for (int nt = 0; nt < 4; ++nt) {
            acc[0][nt] = wmma_bf16(afr[0], bfr[nt], acc[0][nt]);
            acc[1][nt] = wmma_bf16(afr[1], bfr[nt], acc[1][nt]);
        }
    }

    #pragma unroll
    for (int nt = 0; nt < 4; ++nt) {
        const int n = wn + nt * 16 + l15;
        const float bias = b2[n];
        #pragma unroll
        for (int mt = 0; mt < 2; ++mt)
            #pragma unroll
            for (int i = 0; i < 8; ++i) {
                const int m = m0 + wm + mt * 16 + half * 8 + i;
                h2b[(size_t)m * H2sz + n] = f2bf_bits(fmaxf(acc[mt][nt][i] + bias, 0.f));
            }
    }
}

// ---------------------------------------------------------------- fused heads
// out[b,p] = sum_a relu(h2[b,:] @ W3T[p,a,:] + b3[p,a]) * W4[p,a] + b4[p]
// WG: 128 M-rows x 1 protein; 8 waves, each 16(M) x 64(A); 'a' tensor never hits memory.
__global__ __launch_bounds__(256) void k_heads(
    const uint16_t* __restrict__ h2b, const uint16_t* __restrict__ w3t,
    const float* __restrict__ b3, const float* __restrict__ w4,
    const float* __restrict__ b4, float* __restrict__ out)
{
    const int tid  = threadIdx.x;
    const int lane = tid & 31;
    const int wv   = tid >> 5;
    const int half = lane >> 4;
    const int l15  = lane & 15;
    const int p    = blockIdx.y;
    const int mrow0 = blockIdx.x * 128 + wv * 16;

    const uint16_t* pA0 = h2b + (size_t)(mrow0 + l15) * H2sz + half * 8;
    const uint16_t* pB0 = w3t + ((size_t)p * Asz + l15) * H2sz + half * 8;

    const v8f vzero = {0.f,0.f,0.f,0.f,0.f,0.f,0.f,0.f};
    v8f acc[4];
    #pragma unroll
    for (int nt = 0; nt < 4; ++nt) acc[nt] = vzero;

    #pragma unroll
    for (int k0 = 0; k0 < H2sz; k0 += 32) {
        FragAB afr, bfr[4];
        const uint16_t* pA = pA0 + k0;
        afr.q[0] = *(const uint4*)pA;
        afr.q[1] = *(const uint4*)(pA + 16);
        #pragma unroll
        for (int nt = 0; nt < 4; ++nt) {
            const uint16_t* pB = pB0 + (size_t)(nt * 16) * H2sz + k0;
            bfr[nt].q[0] = *(const uint4*)pB;
            bfr[nt].q[1] = *(const uint4*)(pB + 16);
        }
        #pragma unroll
        for (int nt = 0; nt < 4; ++nt)
            acc[nt] = wmma_bf16(afr, bfr[nt], acc[nt]);
    }

    // epilogue: relu(+b3) * W4 reduced over A (N dim = lanes within each 16-lane half)
    float s[8];
    #pragma unroll
    for (int i = 0; i < 8; ++i) s[i] = 0.f;
    #pragma unroll
    for (int nt = 0; nt < 4; ++nt) {
        const int n = nt * 16 + l15;
        const float bias = b3[p * Asz + n];
        const float w4v  = w4[p * Asz + n];
        #pragma unroll
        for (int i = 0; i < 8; ++i)
            s[i] += fmaxf(acc[nt][i] + bias, 0.f) * w4v;
    }
    // butterfly within each 16-lane half (masks 1,2,4,8 stay inside halves)
    #pragma unroll
    for (int mask = 1; mask <= 8; mask <<= 1)
        #pragma unroll
        for (int i = 0; i < 8; ++i)
            s[i] += __shfl_xor(s[i], mask, 32);

    if (l15 == 0) {                        // lanes 0 (M rows 0..7) and 16 (M rows 8..15)
        const float bb = b4[p];
        #pragma unroll
        for (int i = 0; i < 8; ++i)
            out[(size_t)(mrow0 + half * 8 + i) * Psz + p] = s[i] + bb;
    }
}

// ---------------------------------------------------------------- launch

extern "C" void kernel_launch(void* const* d_in, const int* in_sizes, int n_in,
                              void* d_out, int out_size, void* d_ws, size_t ws_size,
                              hipStream_t stream) {
    const float* x  = (const float*)d_in[0];
    const float* W1 = (const float*)d_in[1];
    const float* b1 = (const float*)d_in[2];
    const float* W2 = (const float*)d_in[3];
    const float* b2 = (const float*)d_in[4];
    const float* W3 = (const float*)d_in[5];
    const float* b3 = (const float*)d_in[6];
    const float* W4 = (const float*)d_in[7];
    const float* b4 = (const float*)d_in[8];
    float* out = (float*)d_out;

    // workspace carve-up (all 16B-aligned): ~37.5 MB total
    char* ws = (char*)d_ws;
    uint16_t* w1b = (uint16_t*)(ws);                         // 512*20000*2   = 20,480,000
    uint16_t* w2b = (uint16_t*)(ws + 20480000);              // 256*512*2     =    262,144
    uint16_t* w3t = (uint16_t*)(ws + 20742144);              // 128*64*256*2  =  4,194,304
    uint16_t* h1b = (uint16_t*)(ws + 24936448);              // 8192*512*2    =  8,388,608
    uint16_t* h2b = (uint16_t*)(ws + 33325056);              // 8192*256*2    =  4,194,304

    k_cvt_bf16<<<4096, 256, 0, stream>>>(W1, w1b, H1sz * Gsz);
    k_cvt_bf16<<<512, 256, 0, stream>>>(W2, w2b, H2sz * H1sz);
    k_w3_transpose<<<(Psz * Asz * H2sz) / 256, 256, 0, stream>>>(W3, w3t);

    k_gemm1<<<Bsz / 64, 512, 0, stream>>>(x, w1b, b1, h1b);
    k_gemm2<<<Bsz / 64, 256, 0, stream>>>(h1b, w2b, b2, h2b);
    k_heads<<<dim3(Bsz / 128, Psz), 256, 0, stream>>>(h2b, w3t, b3, W4, b4, out);
}
